// AttentivePooling_16647293240132
// MI455X (gfx1250) — matrix operand order repaired and checked by reading, not compiled, not run on hardware
//
#include <hip/hip_runtime.h>
#include <math.h>

#define B_    32
#define S_    8192
#define C_    256
#define MID_  128
#define ROWS  128                       // rows per chunk/block
#define CHUNKS_PER_BATCH (S_ / ROWS)    // 64
#define NCHUNK (B_ * S_ / ROWS)         // 2048

typedef __attribute__((ext_vector_type(16))) __bf16 v16bf;
typedef __attribute__((ext_vector_type(8)))  float  v8f;
typedef __attribute__((ext_vector_type(4)))  float  v4f;

// ---------------------------------------------------------------------------
// Pass 1: per 128-row chunk — bf16 WMMA GEMM (x·W1), tanh, ·w2 -> scores,
// chunk-local softmax stats (m, e) + unnormalized weighted row-sum partial.
// ---------------------------------------------------------------------------
__global__ __launch_bounds__(256) void ap_pass1(
    const float* __restrict__ x,    // (B*S, C)
    const float* __restrict__ W1,   // (C, MID)
    const float* __restrict__ b1,   // (MID)
    const float* __restrict__ w2,   // (MID)
    float* __restrict__ part,       // (NCHUNK, C)
    float* __restrict__ me)         // (NCHUNK, 2) = {m, e}
{
    // W1 swizzled into v_wmma_f32_16x16x32_bf16 B-fragment layout:
    // [kk(8)][ct(8)][lane(32)][e(16)] ; element = W1[kk*32 + (lane>>4)*8 + (e&7) + (e>>3)*16][ct*16 + (lane&15)]
    __shared__ __attribute__((aligned(32))) __bf16 w1s[8 * 8 * 32 * 16]; // 64 KB
    __shared__ float b1s[MID_];
    __shared__ float w2s[MID_];
    __shared__ float sc[ROWS];      // scores -> exp weights
    __shared__ float red[2];        // {m, e}

    const int tid   = threadIdx.x;
    const int wave  = tid >> 5;
    const int lane  = tid & 31;
    const int chunk = blockIdx.x;
    const long rowBase = (long)chunk * ROWS;

    for (int i = tid; i < MID_; i += 256) { b1s[i] = b1[i]; w2s[i] = w2[i]; }

    // Stage + swizzle + downconvert W1 (f32 global -> bf16 LDS fragment layout)
    for (int i = 0; i < 128; ++i) {
        int f  = i * 256 + tid;              // 0 .. 32767
        int e  = f & 15;
        int ln = (f >> 4) & 31;
        int ct = (f >> 9) & 7;
        int kk = f >> 12;
        int krow = kk * 32 + ((ln >> 4) * 8) + (e & 7) + ((e >> 3) * 16);
        int ncol = ct * 16 + (ln & 15);
        w1s[f] = (__bf16)W1[krow * MID_ + ncol];
    }
    __syncthreads();

    // ---- GEMM: each wave computes a 16x128 tile of h = x_tile * W1 ----
    const int mrow = lane & 15;     // row within 16-row wave tile
    const int kh   = lane >> 4;     // K-half selector for A fragment
    const float* xr = x + (rowBase + wave * 16 + mrow) * C_;

    v8f acc[8] = {};                // 8 N-tiles of 16x16 f32 accumulators

    #pragma unroll
    for (int kk = 0; kk < 8; ++kk) {
        // A fragment (16x32 bf16): lanes 0-15 rows, K = kh*8+0..7 and 16+kh*8+0..7
        const float* p0 = xr + kk * 32 + kh * 8;
        v4f f0 = *(const v4f*)(p0);
        v4f f1 = *(const v4f*)(p0 + 4);
        v4f f2 = *(const v4f*)(p0 + 16);
        v4f f3 = *(const v4f*)(p0 + 20);
        v16bf a;
        #pragma unroll
        for (int q = 0; q < 4; ++q) {
            a[q]      = (__bf16)f0[q];
            a[4 + q]  = (__bf16)f1[q];
            a[8 + q]  = (__bf16)f2[q];
            a[12 + q] = (__bf16)f3[q];
        }
        #pragma unroll
        for (int ct = 0; ct < 8; ++ct) {
            v16bf bf = *(const v16bf*)&w1s[((kk * 8 + ct) * 32 + lane) * 16];
            acc[ct] = __builtin_amdgcn_wmma_f32_16x16x32_bf16(
                false, a, false, bf, (short)0, acc[ct], false, false);
        }
    }

    // ---- epilogue: tanh(h + b1) · w2, reduce over N (lanes) -> 16 scores ----
    const int ncol = lane & 15;     // N within tile (C/D layout: N = lane&15)
    float pr[8];
    #pragma unroll
    for (int r = 0; r < 8; ++r) pr[r] = 0.f;
    #pragma unroll
    for (int ct = 0; ct < 8; ++ct) {
        float bb = b1s[ct * 16 + ncol];
        float ww = w2s[ct * 16 + ncol];
        #pragma unroll
        for (int r = 0; r < 8; ++r)
            pr[r] += tanhf(acc[ct][r] + bb) * ww;
    }
    // C layout: VGPR r holds (M = r + 8*kh, N = lane&15); reduce the 16-lane halves
    #pragma unroll
    for (int r = 0; r < 8; ++r) {
        float s = pr[r];
        #pragma unroll
        for (int off = 1; off < 16; off <<= 1) s += __shfl_xor(s, off, 32);
        if (ncol == 0) sc[wave * 16 + kh * 8 + r] = s;
    }
    __syncthreads();

    // ---- chunk softmax stats over 128 scores (wave 0) ----
    if (wave == 0) {
        float v0 = sc[lane], v1 = sc[lane + 32], v2 = sc[lane + 64], v3 = sc[lane + 96];
        float m = fmaxf(fmaxf(v0, v1), fmaxf(v2, v3));
        #pragma unroll
        for (int off = 1; off < 32; off <<= 1) m = fmaxf(m, __shfl_xor(m, off, 32));
        float e = __expf(v0 - m) + __expf(v1 - m) + __expf(v2 - m) + __expf(v3 - m);
        #pragma unroll
        for (int off = 1; off < 32; off <<= 1) e += __shfl_xor(e, off, 32);
        if (lane == 0) {
            red[0] = m; red[1] = e;
            me[(long)chunk * 2 + 0] = m;
            me[(long)chunk * 2 + 1] = e;
        }
    }
    __syncthreads();
    float m = red[0];
    if (tid < ROWS) sc[tid] = __expf(sc[tid] - m);   // unnormalized weights
    __syncthreads();

    // ---- weighted row sum: thread = channel, rows are cache-hot ----
    float a0 = 0.f;
    const float* xb = x + rowBase * C_;
    #pragma unroll 4
    for (int r = 0; r < ROWS; ++r)
        a0 = fmaf(sc[r], xb[(long)r * C_ + tid], a0);
    part[(long)chunk * C_ + tid] = a0;
}

// ---------------------------------------------------------------------------
// Pass 2: combine 64 chunk partials per batch with global softmax rescale.
// ---------------------------------------------------------------------------
__global__ __launch_bounds__(256) void ap_pass2(
    const float* __restrict__ part,
    const float* __restrict__ me,
    float* __restrict__ out)        // (B, C)
{
    const int b = blockIdx.x;
    const int c = threadIdx.x;
    const float* meb = me + (long)b * CHUNKS_PER_BATCH * 2;

    float M = -INFINITY;
    for (int i = 0; i < CHUNKS_PER_BATCH; ++i) M = fmaxf(M, meb[2 * i]);
    float Z = 0.f;
    for (int i = 0; i < CHUNKS_PER_BATCH; ++i) Z += meb[2 * i + 1] * __expf(meb[2 * i] - M);

    float acc = 0.f;
    const float* pb = part + (long)b * CHUNKS_PER_BATCH * C_;
    for (int i = 0; i < CHUNKS_PER_BATCH; ++i)
        acc = fmaf(pb[(long)i * C_ + c], __expf(meb[2 * i] - M), acc);

    out[b * C_ + c] = acc / Z;
}

extern "C" void kernel_launch(void* const* d_in, const int* in_sizes, int n_in,
                              void* d_out, int out_size, void* d_ws, size_t ws_size,
                              hipStream_t stream) {
    const float* x  = (const float*)d_in[0];
    const float* W1 = (const float*)d_in[1];
    const float* b1 = (const float*)d_in[2];
    const float* w2 = (const float*)d_in[3];
    // d_in[4] = b2 : softmax is shift-invariant, b2 cancels -> unused.

    float* part = (float*)d_ws;                         // NCHUNK*C floats (2 MB)
    float* me   = part + (long)NCHUNK * C_;             // NCHUNK*2 floats

    ap_pass1<<<NCHUNK, 256, 0, stream>>>(x, W1, b1, w2, part, me);
    ap_pass2<<<B_, 256, 0, stream>>>(part, me, (float*)d_out);
}
